// UniversalLinear_55027120996833
// MI455X (gfx1250) — 8-bit matrix operand order repaired and checked by reading, compiled, not run on hardware
//
#include <hip/hip_runtime.h>
#include <stdint.h>

typedef __attribute__((ext_vector_type(8))) int v8i;

#define DK   512
#define EPSF 1e-8f

__global__ __launch_bounds__(256)
void bitnet_wscale(const float* __restrict__ w, float* __restrict__ wscale, int n) {
  __shared__ float red[8];
  const int tid = threadIdx.x;
  float s = 0.f;
  for (int i = tid; i < n; i += 256) s += fabsf(w[i]);
  s += __shfl_xor(s, 1, 32);
  s += __shfl_xor(s, 2, 32);
  s += __shfl_xor(s, 4, 32);
  s += __shfl_xor(s, 8, 32);
  s += __shfl_xor(s, 16, 32);
  if ((tid & 31) == 0) red[tid >> 5] = s;
  __syncthreads();
  if (tid == 0) {
    float t = 0.f;
    #pragma unroll
    for (int i = 0; i < 8; ++i) t += red[i];
    wscale[0] = fmaxf(t / (float)n, EPSF);
  }
}

__global__ __launch_bounds__(256)
void bitnet_wquant(const float* __restrict__ w, const float* __restrict__ wscale,
                   uint32_t* __restrict__ wq, int n4) {
  const int i = blockIdx.x * 256 + threadIdx.x;
  if (i >= n4) return;
  const float inv = 1.0f / wscale[0];
  const float4 v  = ((const float4*)w)[i];
  const int q0 = (int)rintf(fminf(fmaxf(v.x * inv, -1.f), 1.f));
  const int q1 = (int)rintf(fminf(fmaxf(v.y * inv, -1.f), 1.f));
  const int q2 = (int)rintf(fminf(fmaxf(v.z * inv, -1.f), 1.f));
  const int q3 = (int)rintf(fminf(fmaxf(v.w * inv, -1.f), 1.f));
  wq[i] =  (uint32_t)(uint8_t)(int8_t)q0        |
          ((uint32_t)(uint8_t)(int8_t)q1 <<  8) |
          ((uint32_t)(uint8_t)(int8_t)q2 << 16) |
          ((uint32_t)(uint8_t)(int8_t)q3 << 24);
}

__global__ __launch_bounds__(128)
void bitnet_gemm(const float* __restrict__ x,
                 const float* __restrict__ nw,
                 const uint8_t* __restrict__ wq,
                 const float* __restrict__ wscale,
                 float* __restrict__ out) {
  __shared__ alignas(16) uint32_t xq[16 * 128];
  __shared__ float sc[16];

  const int tid = threadIdx.x;
  const long tokBase = (long)blockIdx.x * 16;

  {
    const int t = tid >> 3;
    const int g = tid & 7;
    const float4* xg  = (const float4*)(x + (tokBase + t) * DK);
    const float4* nwg = (const float4*)nw;

    float4 vv[16];
    float sumsq = 0.f;
    #pragma unroll
    for (int i = 0; i < 16; ++i) {
      const int f = g + i * 8;
      const float4 v = xg[f];
      vv[i] = v;
      sumsq += v.x * v.x + v.y * v.y + v.z * v.z + v.w * v.w;
    }
    sumsq += __shfl_xor(sumsq, 1, 32);
    sumsq += __shfl_xor(sumsq, 2, 32);
    sumsq += __shfl_xor(sumsq, 4, 32);
    const float rms = rsqrtf(sumsq * (1.0f / (float)DK) + EPSF);

    float amax = 0.f;
    #pragma unroll
    for (int i = 0; i < 16; ++i) {
      const int f = g + i * 8;
      const float4 w4 = nwg[f];
      float4 xn;
      xn.x = vv[i].x * rms * w4.x;
      xn.y = vv[i].y * rms * w4.y;
      xn.z = vv[i].z * rms * w4.z;
      xn.w = vv[i].w * rms * w4.w;
      vv[i] = xn;
      amax = fmaxf(amax, fmaxf(fmaxf(fabsf(xn.x), fabsf(xn.y)),
                               fmaxf(fabsf(xn.z), fabsf(xn.w))));
    }
    amax = fmaxf(amax, __shfl_xor(amax, 1, 32));
    amax = fmaxf(amax, __shfl_xor(amax, 2, 32));
    amax = fmaxf(amax, __shfl_xor(amax, 4, 32));

    const float scale = fmaxf(amax, EPSF) * (1.0f / 127.0f);
    if (g == 0) sc[t] = scale;
    const float inv = 1.0f / scale;

    #pragma unroll
    for (int i = 0; i < 16; ++i) {
      const int f = g + i * 8;
      const int q0 = (int)rintf(fminf(fmaxf(vv[i].x * inv, -128.f), 127.f));
      const int q1 = (int)rintf(fminf(fmaxf(vv[i].y * inv, -128.f), 127.f));
      const int q2 = (int)rintf(fminf(fmaxf(vv[i].z * inv, -128.f), 127.f));
      const int q3 = (int)rintf(fminf(fmaxf(vv[i].w * inv, -128.f), 127.f));
      xq[t * 128 + f] =  (uint32_t)(uint8_t)(int8_t)q0        |
                        ((uint32_t)(uint8_t)(int8_t)q1 <<  8) |
                        ((uint32_t)(uint8_t)(int8_t)q2 << 16) |
                        ((uint32_t)(uint8_t)(int8_t)q3 << 24);
    }
  }
  __syncthreads();

  const int lane = tid & 31;
  const int wv   = tid >> 5;
  const int m    = lane & 15;
  const int half = lane >> 4;

  v8i a[8];
  #pragma unroll
  for (int kk = 0; kk < 8; ++kk) {
    #pragma unroll
    for (int j = 0; j < 4; ++j) {
      const uint64_t d =
          *(const uint64_t*)&xq[m * 128 + kk * 16 + (j >> 1) * 8 + half * 4 + (j & 1) * 2];
      a[kk][2 * j]     = (int)(uint32_t)(d & 0xffffffffull);
      a[kk][2 * j + 1] = (int)(uint32_t)(d >> 32);
    }
  }

  const float wsc = wscale[0];
  float tsc[8];
  #pragma unroll
  for (int r = 0; r < 8; ++r) tsc[r] = sc[half * 8 + r];

  const int nBase = wv * 128;
  #pragma unroll
  for (int nt = 0; nt < 8; ++nt) {
    const int o = nBase + nt * 16 + m;
    const uint8_t* wrow = wq + (size_t)o * DK;

    v8i acc = {0, 0, 0, 0, 0, 0, 0, 0};
    #pragma unroll
    for (int kk = 0; kk < 8; ++kk) {
      v8i b;
      #pragma unroll
      for (int c = 0; c < 2; ++c) {
        const uint4 q = *(const uint4*)(wrow + kk * 64 + c * 32 + half * 16);
        b[4 * c + 0] = (int)q.x;
        b[4 * c + 1] = (int)q.y;
        b[4 * c + 2] = (int)q.z;
        b[4 * c + 3] = (int)q.w;
      }
      acc = __builtin_amdgcn_wmma_i32_16x16x64_iu8(
           true, a[kk],  true, b, acc,
           false,  false);
    }

    #pragma unroll
    for (int r = 0; r < 8; ++r) {
      const long row = tokBase + half * 8 + r;
      out[row * (long)DK + nBase + nt * 16 + m] = (float)acc[r] * tsc[r] * wsc;
    }
  }
}

extern "C" void kernel_launch(void* const* d_in, const int* in_sizes, int n_in,
                              void* d_out, int out_size, void* d_ws, size_t ws_size,
                              hipStream_t stream) {
  const float* x   = (const float*)d_in[0];
  const float* w   = (const float*)d_in[1];
  const float* nwt = (const float*)d_in[2];
  float* out = (float*)d_out;

  const int nW   = 512 * 512;
  const int M    = in_sizes[0] / DK;
  const int grid = M / 16;

  float*    wscale = (float*)d_ws;
  uint8_t*  wq     = (uint8_t*)d_ws + 256;

  bitnet_wscale<<<1, 256, 0, stream>>>(w, wscale, nW);
  bitnet_wquant<<<nW / 4 / 256, 256, 0, stream>>>(w, wscale, (uint32_t*)wq, nW / 4);
  bitnet_gemm<<<grid, 128, 0, stream>>>(x, nwt, wq, wscale, out);
}
